// GCN_53085795778707
// MI455X (gfx1250) — matrix operand order repaired and checked by reading, not compiled, not run on hardware
//
#include <hip/hip_runtime.h>
#include <hip/hip_bf16.h>
#include <math.h>

// ---------------------------------------------------------------------------
// GCN 2-layer forward for MI455X (gfx1250, wave32, WMMA).
// GEMMs: v_wmma_f32_16x16x32_bf16 with bf16 hi/lo split (3 WMMAs ~ fp32).
// A-fragment fp32->bf16 split is done ONCE per block per k-step, staged in
// double-buffered LDS, shared by all N-tile waves (8x less VALU conversion
// and 8x less HBM traffic on x than per-wave conversion).
// SpMM: b128 gathers + global_atomic_add_f32 (all intermediates L2-resident).
// ---------------------------------------------------------------------------

typedef __attribute__((ext_vector_type(16))) __bf16 v16bf;
typedef __attribute__((ext_vector_type(8)))  float  v8f;

#define N_NODES 50000
#define N_EDGES 1600000
#define NFEAT   512
#define NHID    128
#define NCLASS  40

__device__ __forceinline__ unsigned short f32_to_bf16_rne(float f) {
    union { float f; unsigned int u; } c; c.f = f;
    unsigned int u = c.u;
    u += 0x7FFFu + ((u >> 16) & 1u);        // round-to-nearest-even
    return (unsigned short)(u >> 16);
}
__device__ __forceinline__ float bf16_bits_to_f32(unsigned short h) {
    union { unsigned int u; float f; } c; c.u = ((unsigned int)h) << 16;
    return c.f;
}

// ---------------------------------------------------------------------------
// Pack a [Kdim, Ncols] row-major fp32 weight matrix into WMMA B-fragment
// layout, split into bf16 hi/lo. Buffer layout: [nt][kt][lane][e] bf16.
// B 32x16 bf16 layout: lane = 16*grp + n ; element e -> K = 32*kt + 8*grp + (e<8 ? e : e+8)
// ---------------------------------------------------------------------------
__global__ void pack_B_kernel(const float* __restrict__ W,
                              unsigned short* __restrict__ bhi,
                              unsigned short* __restrict__ blo,
                              int Ncols, int nTiles, int kTiles) {
    int idx = blockIdx.x * blockDim.x + threadIdx.x;
    int total = nTiles * kTiles * 32 * 16;
    if (idx >= total) return;
    int e    = idx & 15;
    int lane = (idx >> 4) & 31;
    int kt   = (idx >> 9) % kTiles;
    int nt   = idx / (kTiles << 9);
    int grp  = lane >> 4;
    int col  = nt * 16 + (lane & 15);
    int K    = kt * 32 + grp * 8 + (e < 8 ? e : e + 8);
    float v  = (col < Ncols) ? W[(size_t)K * Ncols + col] : 0.0f;
    unsigned short hi = f32_to_bf16_rne(v);
    unsigned short lo = f32_to_bf16_rne(v - bf16_bits_to_f32(hi));
    bhi[idx] = hi;
    blo[idx] = lo;
}

// ---------------------------------------------------------------------------
// WMMA GEMM with LDS-staged A fragments.
// grid.x = Mtiles; blockDim.x = 32*nTiles (wave w -> N-tile w).
// Per k-step: all threads cooperatively convert the 512 A-fragment slots
// (16x32 fp32 tile) to bf16 hi/lo in LDS (A-layout: lanes 0-15 K=8g+e / e<8,
// 16+8g+(e-8) / e>=8), one barrier, then each wave ds-loads its v16bf pair
// and issues 3 WMMAs (hi*hi, hi*lo, lo*hi).
// ---------------------------------------------------------------------------
__global__ void wmma_gemm_lds(const float* __restrict__ A,
                              const unsigned short* __restrict__ Bhi,
                              const unsigned short* __restrict__ Blo,
                              float* __restrict__ C,
                              int nTiles, int kTiles, int lda, int Ncols) {
    __shared__ alignas(32) unsigned short sAhi[2][512];   // [buf][lane*16+e]
    __shared__ alignas(32) unsigned short sAlo[2][512];

    const int tm   = blockIdx.x;
    const int tn   = threadIdx.x >> 5;
    const int lane = threadIdx.x & 31;

    v8f acc = {};
    for (int kt = 0; kt < kTiles; ++kt) {
        const int buf = kt & 1;
        // ---- cooperative A conversion into LDS (ISA A-fragment layout)
        for (int slot = threadIdx.x; slot < 512; slot += blockDim.x) {
            int l   = slot >> 4;
            int e   = slot & 15;
            int grp = l >> 4;
            int row = tm * 16 + (l & 15);
            int K   = kt * 32 + 8 * grp + (e < 8 ? e : e + 8);
            float v = A[(size_t)row * lda + K];
            unsigned short h = f32_to_bf16_rne(v);
            sAhi[buf][slot] = h;
            sAlo[buf][slot] = f32_to_bf16_rne(v - bf16_bits_to_f32(h));
        }
        __syncthreads();   // single barrier per step (double-buffered LDS)

        v16bf Ah = *((const v16bf*)&sAhi[buf][0] + lane);
        v16bf Al = *((const v16bf*)&sAlo[buf][0] + lane);
        size_t boff = ((size_t)(tn * kTiles + kt) * 32 + lane);
        v16bf Bh = ((const v16bf*)Bhi)[boff];
        v16bf Bl = ((const v16bf*)Blo)[boff];

        acc = __builtin_amdgcn_wmma_f32_16x16x32_bf16(false, Ah, false, Bh,
                                                      (short)0, acc, false, false);
        acc = __builtin_amdgcn_wmma_f32_16x16x32_bf16(false, Ah, false, Bl,
                                                      (short)0, acc, false, false);
        acc = __builtin_amdgcn_wmma_f32_16x16x32_bf16(false, Al, false, Bh,
                                                      (short)0, acc, false, false);
    }

    // ---- C fragment store: VGPR r -> row = 16*tm + 8*grp + r, col = lane&15
    const int grp = lane >> 4;
    const int col = tn * 16 + (lane & 15);
    if (col < Ncols) {
        #pragma unroll
        for (int r = 0; r < 8; ++r) {
            int row = tm * 16 + grp * 8 + r;
            C[(size_t)row * Ncols + col] = acc[r];
        }
    }
}

// ---------------------------------------------------------------------------
// SpMM scatter, 128 features: one wave per edge, b128 gather, 4 f32 atomics.
// ---------------------------------------------------------------------------
__global__ void spmm_scatter128(const long long* __restrict__ ei,
                                const float* __restrict__ ew,
                                const float* __restrict__ S,
                                float* __restrict__ agg, int E) {
    int e = blockIdx.x * (blockDim.x >> 5) + (threadIdx.x >> 5);
    if (e >= E) return;
    int lane = threadIdx.x & 31;
    long long src = ei[e];
    long long dst = ei[(size_t)E + e];
    float w = ew[e];
    float4 v = *((const float4*)(S + (size_t)src * NHID) + lane);
    float* d = agg + (size_t)dst * NHID + lane * 4;
    atomicAdd(d + 0, w * v.x);
    atomicAdd(d + 1, w * v.y);
    atomicAdd(d + 2, w * v.z);
    atomicAdd(d + 3, w * v.w);
}

// ---------------------------------------------------------------------------
// SpMM scatter, 40 features: one wave per edge; lanes cover f and f+32.
// ---------------------------------------------------------------------------
__global__ void spmm_scatter40(const long long* __restrict__ ei,
                               const float* __restrict__ ew,
                               const float* __restrict__ S,
                               float* __restrict__ agg, int E) {
    int e = blockIdx.x * (blockDim.x >> 5) + (threadIdx.x >> 5);
    if (e >= E) return;
    int lane = threadIdx.x & 31;
    long long src = ei[e];
    long long dst = ei[(size_t)E + e];
    float w = ew[e];
    const float* s = S + (size_t)src * NCLASS;
    float* d = agg + (size_t)dst * NCLASS;
    atomicAdd(d + lane, w * s[lane]);                 // lane < 32 < 40 always
    int f2 = lane + 32;
    if (f2 < NCLASS) atomicAdd(d + f2, w * s[f2]);
}

// ---------------------------------------------------------------------------
// h = relu(h + b[. % 128]) in place, float4-vectorized (128 % 4 == 0).
// ---------------------------------------------------------------------------
__global__ void bias_relu128(float* __restrict__ h, const float* __restrict__ b, int n4) {
    int i = blockIdx.x * blockDim.x + threadIdx.x;
    if (i >= n4) return;
    float4 v = ((const float4*)h)[i];
    float4 bb = *(const float4*)(b + ((i * 4) & (NHID - 1)));
    v.x = fmaxf(v.x + bb.x, 0.0f);
    v.y = fmaxf(v.y + bb.y, 0.0f);
    v.z = fmaxf(v.z + bb.z, 0.0f);
    v.w = fmaxf(v.w + bb.w, 0.0f);
    ((float4*)h)[i] = v;
}

__global__ void zero_f128(float4* __restrict__ p, int n4) {
    int i = blockIdx.x * blockDim.x + threadIdx.x;
    if (i < n4) p[i] = make_float4(0.f, 0.f, 0.f, 0.f);
}

// ---------------------------------------------------------------------------
// Row-wise log_softmax over 40 classes (+bias): one wave per node, shfl_xor
// reductions across the wave32.
// ---------------------------------------------------------------------------
__global__ void logsoftmax40(const float* __restrict__ agg,
                             const float* __restrict__ b,
                             float* __restrict__ out, int N) {
    int node = blockIdx.x * (blockDim.x >> 5) + (threadIdx.x >> 5);
    if (node >= N) return;
    int lane = threadIdx.x & 31;
    const float* row = agg + (size_t)node * NCLASS;
    float v0 = row[lane] + b[lane];
    float v1 = (lane < NCLASS - 32) ? (row[lane + 32] + b[lane + 32]) : -INFINITY;
    float m = fmaxf(v0, v1);
    #pragma unroll
    for (int o = 16; o > 0; o >>= 1) m = fmaxf(m, __shfl_xor(m, o, 32));
    float s = expf(v0 - m) + ((lane < NCLASS - 32) ? expf(v1 - m) : 0.0f);
    #pragma unroll
    for (int o = 16; o > 0; o >>= 1) s += __shfl_xor(s, o, 32);
    float lse = m + logf(s);
    float* orow = out + (size_t)node * NCLASS;
    orow[lane] = v0 - lse;
    if (lane < NCLASS - 32) orow[lane + 32] = v1 - lse;
}

// ---------------------------------------------------------------------------
extern "C" void kernel_launch(void* const* d_in, const int* in_sizes, int n_in,
                              void* d_out, int out_size, void* d_ws, size_t ws_size,
                              hipStream_t stream) {
    (void)in_sizes; (void)n_in; (void)out_size; (void)ws_size;

    const float*     x   = (const float*)d_in[0];
    const long long* ei1 = (const long long*)d_in[2];
    const float*     ew1 = (const float*)d_in[3];
    const long long* ei2 = (const long long*)d_in[4];
    const float*     ew2 = (const float*)d_in[5];
    const float*     W1  = (const float*)d_in[6];
    const float*     b1  = (const float*)d_in[7];
    const float*     W2  = (const float*)d_in[8];
    const float*     b2  = (const float*)d_in[9];
    float* out = (float*)d_out;

    // Workspace carve-up (all offsets 256B-aligned)
    char* ws = (char*)d_ws;
    float*          support1 = (float*)(ws + 0);                    // 50000*128*4 = 25,600,000
    float*          agg1     = (float*)(ws + 25600000);             // 25,600,000
    float*          support2 = (float*)(ws + 51200000);             // 50000*40*4 =  8,000,000
    float*          agg2     = (float*)(ws + 59200000);             //  8,000,000
    unsigned short* w1bhi    = (unsigned short*)(ws + 67200000);    // 8*16*512*2 = 131,072
    unsigned short* w1blo    = (unsigned short*)(ws + 67331072);    // 131,072
    unsigned short* w2bhi    = (unsigned short*)(ws + 67462144);    // 3*4*512*2 = 12,288
    unsigned short* w2blo    = (unsigned short*)(ws + 67474432);    // 12,288

    const int mtiles = N_NODES / 16;                     // 3125
    const int nt1 = NHID / 16,          kt1 = NFEAT / 32; // 8, 16
    const int nt2 = (NCLASS + 15) / 16, kt2 = NHID / 32;  // 3, 4

    // 1) Pack W1 / W2 into WMMA B-fragment layout (bf16 hi/lo)
    pack_B_kernel<<<(nt1 * kt1 * 512 + 255) / 256, 256, 0, stream>>>(
        W1, w1bhi, w1blo, NHID, nt1, kt1);
    pack_B_kernel<<<(nt2 * kt2 * 512 + 255) / 256, 256, 0, stream>>>(
        W2, w2bhi, w2blo, NCLASS, nt2, kt2);

    // 2) support1 = x @ W1   (WMMA, bf16x3, LDS-staged A; 8 waves/block)
    wmma_gemm_lds<<<mtiles, 32 * nt1, 0, stream>>>(
        x, w1bhi, w1blo, support1, nt1, kt1, NFEAT, NHID);

    // 3) agg1 = segment_sum(w * support1[src])
    zero_f128<<<(N_NODES * NHID / 4 + 255) / 256, 256, 0, stream>>>(
        (float4*)agg1, N_NODES * NHID / 4);
    spmm_scatter128<<<(N_EDGES + 7) / 8, 256, 0, stream>>>(
        ei1, ew1, support1, agg1, N_EDGES);

    // 4) h1 = relu(agg1 + b1)  (in place)
    bias_relu128<<<(N_NODES * NHID / 4 + 255) / 256, 256, 0, stream>>>(
        agg1, b1, N_NODES * NHID / 4);

    // 5) support2 = h1 @ W2   (WMMA, bf16x3, N padded 40 -> 48 w/ store guard)
    wmma_gemm_lds<<<mtiles, 32 * nt2, 0, stream>>>(
        agg1, w2bhi, w2blo, support2, nt2, kt2, NHID, NCLASS);

    // 6) agg2 = segment_sum(w * support2[src])
    zero_f128<<<(N_NODES * NCLASS / 4 + 255) / 256, 256, 0, stream>>>(
        (float4*)agg2, N_NODES * NCLASS / 4);
    spmm_scatter40<<<(N_EDGES + 7) / 8, 256, 0, stream>>>(
        ei2, ew2, support2, agg2, N_EDGES);

    // 7) out = log_softmax(agg2 + b2)
    logsoftmax40<<<(N_NODES + 7) / 8, 256, 0, stream>>>(agg2, b2, out, N_NODES);
}
